// PoleAgent003_72138270703717
// MI455X (gfx1250) — matrix operand order repaired and checked by reading, not compiled, not run on hardware
//
#include <hip/hip_runtime.h>
#include <hip/hip_bf16.h>
#include <stdint.h>
#include <stddef.h>

#define PI_F 3.14159265358979323846f

#define NHRZ   2048
#define NB     128
#define NUU    64
#define NYY    64
#define NHH    256
#define HHALF  128
#define LCH    64            // chunk length (time steps per workgroup)
#define NCH    (NHRZ / LCH)  // 32 chunks
#define BT     16            // batch rows per workgroup
#define NBT    (NB / BT)     // 8 batch tiles

typedef __attribute__((ext_vector_type(16))) __bf16 v16bf;
typedef __attribute__((ext_vector_type(8)))  float  v8f;

struct Frag { uint32_t d[8]; };

static __device__ __forceinline__ v16bf frag_bf(const Frag& f) {
  return __builtin_bit_cast(v16bf, f);
}
static __device__ __forceinline__ uint32_t pk_bf16(float x, float y) {
  __bf16 a = (__bf16)x, b = (__bf16)y;
  uint16_t ua = __builtin_bit_cast(uint16_t, a);
  uint16_t ub = __builtin_bit_cast(uint16_t, b);
  return ((uint32_t)ub << 16) | (uint32_t)ua;
}
static __device__ __forceinline__ v8f wmma_bf16(const Frag& a, const Frag& b, v8f c) {
  // D = A(16x32 bf16) * B(32x16 bf16) + C(16x16 f32)
  return __builtin_amdgcn_wmma_f32_16x16x32_bf16(false, frag_bf(a), false, frag_bf(b),
                                                 (short)0, c, false, false);
}

// --- CDNA5 async global->LDS copy (tracked by ASYNCcnt, not LOADcnt/DScnt) ---
static __device__ __forceinline__ uint32_t lds_off32(const void* p) {
  // generic pointer to __shared__ = {shared aperture, lds_offset[31:0]}
  return (uint32_t)(uintptr_t)p;
}
static __device__ __forceinline__ void async_copy_b128(uint32_t lds_dst,
                                                       const void* gsrc) {
  asm volatile("global_load_async_to_lds_b128 %0, %1, off"
               :: "v"(lds_dst), "v"((uint64_t)(uintptr_t)gsrc)
               : "memory");
}
static __device__ __forceinline__ void wait_async0() {
#if defined(__has_builtin)
#if __has_builtin(__builtin_amdgcn_s_wait_asynccnt)
  __builtin_amdgcn_s_wait_asynccnt(0);
#else
  asm volatile("s_wait_asynccnt 0x0" ::: "memory");
#endif
#else
  asm volatile("s_wait_asynccnt 0x0" ::: "memory");
#endif
}

// ---------------------------------------------------------------------------
// Kernel 0: precompute lr/li tables, bf16 B*nf (GEMM1 weights), bf16 Wx2y^T.
// ---------------------------------------------------------------------------
__global__ __launch_bounds__(256) void k_prep(const float* __restrict__ lmr,
                                              const float* __restrict__ lmi,
                                              const float* __restrict__ B,
                                              const float* __restrict__ Wx2y,
                                              uint16_t* __restrict__ Bs,
                                              uint16_t* __restrict__ W2t,
                                              float* __restrict__ lrT,
                                              float* __restrict__ liT) {
  int h = threadIdx.x;              // 0..255
  int k = h & (HHALF - 1);
  float ar = fabsf(lmr[k]);
  float r  = expf(-ar);
  float th = 0.5f * PI_F * lmi[k];
  float lr = r * cosf(th), li = r * sinf(th);
  float nf = sqrtf(fmaxf(0.f, 1.f - expf(-2.f * ar)));
  if (h < HHALF) { lrT[h] = lr; liT[h] = li; }
  for (int u = 0; u < NUU; ++u) {
    __bf16 v = (__bf16)(B[h * NUU + u] * nf);
    Bs[h * NUU + u] = __builtin_bit_cast(uint16_t, v);
  }
  for (int y = 0; y < NYY; ++y) {
    __bf16 v = (__bf16)(Wx2y[y * NHH + h]);   // transpose: W2t[h][y]
    W2t[h * NYY + y] = __builtin_bit_cast(uint16_t, v);
  }
}

// ---------------------------------------------------------------------------
// Kernel A: per-chunk local scan from zero state; stores only the final state.
// grid = NCH * NBT blocks, 256 threads (8 waves).
// ---------------------------------------------------------------------------
__global__ __launch_bounds__(256) void k_chunkA(const float* __restrict__ U,
                                                const uint16_t* __restrict__ Bs,
                                                const float* __restrict__ lrT,
                                                const float* __restrict__ liT,
                                                float* __restrict__ Tout) {
  __shared__ float ldsU[2][BT * 68];   // padded stride 68 (bank-friendly)

  const int tid  = threadIdx.x;
  const int wave = tid >> 5, lane = tid & 31, ln = lane & 15, hi = lane >> 4;
  const int cb   = wave * 16;                  // this wave's column base (0..112)
  const int chunk = blockIdx.x / NBT;
  const int b0    = (blockIdx.x % NBT) * BT;

  const float lr = lrT[cb + ln];
  const float li = liT[cb + ln];

  // Persistent GEMM1 B-fragments: Bs columns for this wave's two h-tiles.
  Frag bfr[2][2];   // [tile][ki]  (K = u, split 0..31 / 32..63)
  for (int t2 = 0; t2 < 2; ++t2)
    for (int ki = 0; ki < 2; ++ki)
      for (int v = 0; v < 8; ++v) {
        int u = ki * 32 + (hi ? 8 : 0) + (v < 4 ? 2 * v : 16 + 2 * (v - 4));
        int h = t2 * HHALF + cb + ln;
        const uint16_t* p = &Bs[h * NUU + u];
        bfr[t2][ki].d[v] = (uint32_t)p[0] | ((uint32_t)p[1] << 16);
      }

  v8f c0 = {}, c1 = {};   // state fragments: x1 cols, x2 cols (f32)

  const int m0 = tid >> 4, q0 = tid & 15;
  const char* __restrict__ gbase =
      (const char*)&U[((size_t)(chunk * LCH) * NB + b0) * NUU];
  const uint32_t ldst0 = lds_off32(&ldsU[0][m0 * 68 + q0 * 4]);
  const uint32_t ldst1 = lds_off32(&ldsU[1][m0 * 68 + q0 * 4]);

  // async preload of tile 0 (each wave copies 512B; ASYNCcnt-tracked)
  async_copy_b128(ldst0, gbase + (size_t)tid * 16);
  wait_async0();
  __syncthreads();

  for (int j = 0; j < LCH; ++j) {
    const bool has = (j + 1 < LCH);
    if (has)
      async_copy_b128(((j + 1) & 1) ? ldst1 : ldst0,
                      gbase + (size_t)(j + 1) * (NB * NUU * 4) + (size_t)tid * 16);

    // rotate state (per-lane 2x2 rotation)
    v8f n0, n1;
#pragma unroll
    for (int i = 0; i < 8; ++i) {
      n0[i] = lr * c0[i] - li * c1[i];
      n1[i] = li * c0[i] + lr * c1[i];
    }
    // A fragments from the staged U tile (f32 -> packed bf16)
    Frag a[2];
    const float* ub = &ldsU[j & 1][0];
#pragma unroll
    for (int ki = 0; ki < 2; ++ki)
#pragma unroll
      for (int v = 0; v < 8; ++v) {
        int k = (v < 4 ? 2 * v : 16 + 2 * (v - 4)) + (hi ? 8 : 0) + ki * 32;
        float2 f = *(const float2*)&ub[ln * 68 + k];
        a[ki].d[v] = pk_bf16(f.x, f.y);
      }
    // accumulate Bu into rotated state:  x' = R x + U B^T
    n0 = wmma_bf16(a[0], bfr[0][0], n0);
    n0 = wmma_bf16(a[1], bfr[0][1], n0);
    n1 = wmma_bf16(a[0], bfr[1][0], n1);
    n1 = wmma_bf16(a[1], bfr[1][1], n1);
    c0 = n0; c1 = n1;

    if (has) wait_async0();   // next tile landed (overlapped with compute above)
    __syncthreads();
  }

  // store chunk-total state T_c
#pragma unroll
  for (int r = 0; r < 8; ++r) {
    int b = b0 + r + hi * 8;
    size_t base = ((size_t)chunk * NB + b) * NHH;
    Tout[base + cb + ln]         = c0[r];
    Tout[base + HHALF + cb + ln] = c1[r];
  }
}

// ---------------------------------------------------------------------------
// Kernel B: x0 = y0 @ Wy2x^T + by2x; chunk-prefix propagation with closed-form
// R(L) = r^L * rot(L*theta).  One thread per (batch, pair).
// ---------------------------------------------------------------------------
__global__ __launch_bounds__(256) void k_prefix(const float* __restrict__ y0,
                                                const float* __restrict__ lmr,
                                                const float* __restrict__ lmi,
                                                const float* __restrict__ Wy2x,
                                                const float* __restrict__ by2x,
                                                const float* __restrict__ Tin,
                                                float* __restrict__ Pout) {
  int id = blockIdx.x * 256 + threadIdx.x;   // 0..16383
  int b = id >> 7, k = id & 127;
  float ar = fabsf(lmr[k]);
  float th = 0.5f * PI_F * lmi[k];
  float rL = expf(-(float)LCH * ar);
  float aL = (float)LCH * th;
  float cl = rL * cosf(aL), sl = rL * sinf(aL);

  float s1 = by2x[k], s2 = by2x[k + HHALF];
  for (int n = 0; n < NYY; ++n) {
    float yv = y0[b * NYY + n];
    s1 += yv * Wy2x[k * NYY + n];
    s2 += yv * Wy2x[(k + HHALF) * NYY + n];
  }
  for (int c = 0; c < NCH; ++c) {
    size_t base = ((size_t)c * NB + b) * NHH;
    Pout[base + k]         = s1;
    Pout[base + k + HHALF] = s2;
    if (c + 1 < NCH) {
      float t1 = Tin[base + k], t2 = Tin[base + k + HHALF];
      float ns1 = cl * s1 - sl * s2 + t1;
      float ns2 = sl * s1 + cl * s2 + t2;
      s1 = ns1; s2 = ns2;
    }
  }
}

// ---------------------------------------------------------------------------
// Kernel C: re-run each chunk from its true prefix state, fused with the
// output projection Y = x @ Wx2y^T + b (per-wave K-partial WMMA + LDS reduce).
// ---------------------------------------------------------------------------
__global__ __launch_bounds__(256) void k_chunkC(const float* __restrict__ U,
                                                const uint16_t* __restrict__ Bs,
                                                const uint16_t* __restrict__ W2t,
                                                const float* __restrict__ lrT,
                                                const float* __restrict__ liT,
                                                const float* __restrict__ Pin,
                                                const float* __restrict__ bx2y,
                                                float* __restrict__ Y) {
  __shared__ float ldsU[2][BT * 68];   // 8704 B
  __shared__ float xbuf[8][16 * 34];   // wave-private transpose bounce, 17408 B
  __shared__ float ybuf[8][16 * 68];   // per-wave Y partials, 34816 B

  const int tid  = threadIdx.x;
  const int wave = tid >> 5, lane = tid & 31, ln = lane & 15, hi = lane >> 4;
  const int cb   = wave * 16;
  const int chunk = blockIdx.x / NBT;
  const int b0    = (blockIdx.x % NBT) * BT;

  const float lr = lrT[cb + ln];
  const float li = liT[cb + ln];

  Frag bfr[2][2];
  for (int t2 = 0; t2 < 2; ++t2)
    for (int ki = 0; ki < 2; ++ki)
      for (int v = 0; v < 8; ++v) {
        int u = ki * 32 + (hi ? 8 : 0) + (v < 4 ? 2 * v : 16 + 2 * (v - 4));
        int h = t2 * HHALF + cb + ln;
        const uint16_t* p = &Bs[h * NUU + u];
        bfr[t2][ki].d[v] = (uint32_t)p[0] | ((uint32_t)p[1] << 16);
      }

  // Persistent GEMM2 B-fragments: Wx2y^T rows for this wave's local K order.
  Frag wf[4];   // 4 y-tiles of 16
  for (int t2 = 0; t2 < 4; ++t2)
    for (int v = 0; v < 8; ++v) {
      int h0 = (v < 4 ? cb + 2 * v : HHALF + cb + 2 * (v - 4)) + (hi ? 8 : 0);   // local k -> h map
      int y  = t2 * 16 + ln;
      wf[t2].d[v] = (uint32_t)W2t[h0 * NYY + y] |
                    ((uint32_t)W2t[(h0 + 1) * NYY + y] << 16);
    }

  // initial state = P_chunk
  v8f c0, c1;
#pragma unroll
  for (int r = 0; r < 8; ++r) {
    size_t base = ((size_t)chunk * NB + b0 + r + hi * 8) * NHH;
    c0[r] = Pin[base + cb + ln];
    c1[r] = Pin[base + HHALF + cb + ln];
  }
  const float4 bias = *(const float4*)&bx2y[(tid & 15) * 4];

  const int m0 = tid >> 4, q0 = tid & 15;
  const char* __restrict__ gbase =
      (const char*)&U[((size_t)(chunk * LCH) * NB + b0) * NUU];
  const uint32_t ldst0 = lds_off32(&ldsU[0][m0 * 68 + q0 * 4]);
  const uint32_t ldst1 = lds_off32(&ldsU[1][m0 * 68 + q0 * 4]);

  auto emit_y = [&](int t) {
    // bounce state through LDS to build the (transposed) A-fragment
#pragma unroll
    for (int r = 0; r < 8; ++r) {
      int m = r + hi * 8;
      xbuf[wave][m * 34 + ln]      = c0[r];
      xbuf[wave][m * 34 + 16 + ln] = c1[r];
    }
    Frag a2;
#pragma unroll
    for (int v = 0; v < 8; ++v) {
      int k2 = (v < 4 ? 2 * v : 16 + 2 * (v - 4)) + (hi ? 8 : 0);
      float2 f = *(const float2*)&xbuf[wave][ln * 34 + k2];
      a2.d[v] = pk_bf16(f.x, f.y);
    }
#pragma unroll
    for (int t2 = 0; t2 < 4; ++t2) {
      v8f z = {};
      v8f yt = wmma_bf16(a2, wf[t2], z);     // partial over this wave's K=32
#pragma unroll
      for (int r = 0; r < 8; ++r) {
        int m = r + hi * 8;
        ybuf[wave][m * 68 + t2 * 16 + ln] = yt[r];
      }
    }
    __syncthreads();
    {
      int m = tid >> 4, yg = (tid & 15) * 4;
      float4 acc = bias;
#pragma unroll
      for (int w = 0; w < 8; ++w) {
        float4 p = *(const float4*)&ybuf[w][m * 68 + yg];
        acc.x += p.x; acc.y += p.y; acc.z += p.z; acc.w += p.w;
      }
      *(float4*)&Y[(((size_t)t) * NB + b0 + m) * NYY + yg] = acc;
    }
    __syncthreads();
  };

  // async preload of tile 0
  async_copy_b128(ldst0, gbase + (size_t)tid * 16);
  wait_async0();
  __syncthreads();

  if (chunk == 0) emit_y(0);   // Y[0] from x0 (uniform branch per block)

  for (int j = 0; j < LCH; ++j) {
    const bool has = (j + 1 < LCH);
    if (has)
      async_copy_b128(((j + 1) & 1) ? ldst1 : ldst0,
                      gbase + (size_t)(j + 1) * (NB * NUU * 4) + (size_t)tid * 16);

    v8f n0, n1;
#pragma unroll
    for (int i = 0; i < 8; ++i) {
      n0[i] = lr * c0[i] - li * c1[i];
      n1[i] = li * c0[i] + lr * c1[i];
    }
    Frag a[2];
    const float* ub = &ldsU[j & 1][0];
#pragma unroll
    for (int ki = 0; ki < 2; ++ki)
#pragma unroll
      for (int v = 0; v < 8; ++v) {
        int k = (v < 4 ? 2 * v : 16 + 2 * (v - 4)) + (hi ? 8 : 0) + ki * 32;
        float2 f = *(const float2*)&ub[ln * 68 + k];
        a[ki].d[v] = pk_bf16(f.x, f.y);
      }
    n0 = wmma_bf16(a[0], bfr[0][0], n0);
    n0 = wmma_bf16(a[1], bfr[0][1], n0);
    n1 = wmma_bf16(a[0], bfr[1][0], n1);
    n1 = wmma_bf16(a[1], bfr[1][1], n1);
    c0 = n0; c1 = n1;

    if (has) wait_async0();        // next U tile landed during compute
    emit_y(chunk * LCH + j + 1);   // barriers inside also publish next U tile
  }
}

// ---------------------------------------------------------------------------
extern "C" void kernel_launch(void* const* d_in, const int* in_sizes, int n_in,
                              void* d_out, int out_size, void* d_ws, size_t ws_size,
                              hipStream_t stream) {
  (void)in_sizes; (void)n_in; (void)out_size; (void)ws_size;
  const float* y0   = (const float*)d_in[0];
  const float* U    = (const float*)d_in[1];
  const float* lmr  = (const float*)d_in[2];
  const float* lmi  = (const float*)d_in[3];
  const float* B    = (const float*)d_in[4];
  const float* Wy2x = (const float*)d_in[5];
  const float* by2x = (const float*)d_in[6];
  const float* Wx2y = (const float*)d_in[7];
  const float* bx2y = (const float*)d_in[8];
  float* Y = (float*)d_out;

  char* ws = (char*)d_ws;
  uint16_t* Bs  = (uint16_t*)(ws);                    // 32 KB
  uint16_t* W2t = (uint16_t*)(ws + 32768);            // 32 KB
  float*    lrT = (float*)(ws + 65536);               // 512 B
  float*    liT = (float*)(ws + 65536 + 512);         // 512 B
  float*    Tt  = (float*)(ws + 66560);               // 4 MB
  float*    P   = (float*)(ws + 66560 + (size_t)NCH * NB * NHH * 4);  // 4 MB

  k_prep  <<<1,            256, 0, stream>>>(lmr, lmi, B, Wx2y, Bs, W2t, lrT, liT);
  k_chunkA<<<NCH * NBT,    256, 0, stream>>>(U, Bs, lrT, liT, Tt);
  k_prefix<<<(NB * HHALF) / 256, 256, 0, stream>>>(y0, lmr, lmi, Wy2x, by2x, Tt, P);
  k_chunkC<<<NCH * NBT,    256, 0, stream>>>(U, Bs, W2t, lrT, liT, P, bx2y, Y);
}
